// GRU_EncoderCell_58634893525362
// MI455X (gfx1250) — compile-verified, matrix operand order checked by reading, not run on hardware
//
#include <hip/hip_runtime.h>
#include <hip/hip_bf16.h>

typedef __attribute__((ext_vector_type(16))) __bf16 bf16x16;
typedef __attribute__((ext_vector_type(8)))  __bf16 bf16x8;
typedef __attribute__((ext_vector_type(8)))  float  f32x8;

#define B_  512
#define T_  49
#define I_  96
#define H_  1024
#define G3_ 3072   // 3*H

// Load one 16x32 bf16 A/B fragment row slice for this lane.
// Per ISA 7.12.2: lanes 0-15 hold K {0..7,16..23}, lanes 16-31 hold K {8..15,24..31},
// packed 2 consecutive K per VGPR -> two 16-byte loads per lane.
__device__ __forceinline__ bf16x16 load_frag(const __bf16* __restrict__ rowPtr,
                                             int kt, int half) {
  const bf16x8 lo = *(const bf16x8*)(rowPtr + kt + half * 8);
  const bf16x8 hi = *(const bf16x8*)(rowPtr + kt + 16 + half * 8);
  return __builtin_shufflevector(lo, hi, 0, 1, 2, 3, 4, 5, 6, 7,
                                         8, 9, 10, 11, 12, 13, 14, 15);
}

// Gate GEMM: blockIdx.x -> 64-row M tile (8 tiles over B=512)
//            blockIdx.y -> 16 column groups of 256 over a virtual 4096-wide N:
//              [0,4)  : r  = sigmoid(x@Wih_r^T + h@Whh_r^T + b)
//              [4,8)  : z  = sigmoid(x@Wih_z^T + h@Whh_z^T + b)
//              [8,12) : hn = h@Whh_n^T + b_hh_n          (kept raw)
//              [12,16): inn= x@Wih_n^T + b_ih_n          (kept raw)
__global__ __launch_bounds__(256)
void gru_gate_gemm(const __bf16* __restrict__ Whh,   // [3072,1024] row-major bf16
                   const __bf16* __restrict__ Wih,   // [3072,96]   row-major bf16
                   const __bf16* __restrict__ Enc,   // [512,49,96] bf16
                   const __bf16* __restrict__ Hb,    // [512,1024]  bf16 hidden state
                   const float*  __restrict__ b_ih,
                   const float*  __restrict__ b_hh,
                   float* __restrict__ Rg, float* __restrict__ Zg,
                   float* __restrict__ HN, float* __restrict__ INN,
                   int t) {
  const int tid   = threadIdx.x;
  const int lane  = tid & 31;
  const int half  = lane >> 4;
  const int lm    = lane & 15;
  const int wave  = tid >> 5;
  const int waveM = wave >> 2;  // 0..1
  const int waveN = wave & 3;   // 0..3

  const int m0   = blockIdx.x * 64 + waveM * 32;  // wave row base in [0,512)
  const int y    = blockIdx.y;
  const int part = y >> 2;                        // 0=r 1=z 2=hn 3=inn (block-uniform)
  const int n0   = (y & 3) * 256 + waveN * 64;    // wave col base in [0,1024)

  f32x8 acc[2][4];
#pragma unroll
  for (int i = 0; i < 2; ++i)
#pragma unroll
    for (int j = 0; j < 4; ++j)
#pragma unroll
      for (int r = 0; r < 8; ++r) acc[i][j][r] = 0.0f;

  // ---- hidden contribution (parts r, z, hn) : K = 1024 ----
  if (part < 3) {
    const __bf16* aRow0 = Hb + (size_t)(m0 + lm) * H_;
    const __bf16* aRow1 = Hb + (size_t)(m0 + 16 + lm) * H_;
    const __bf16* bRow[4];
#pragma unroll
    for (int j = 0; j < 4; ++j)
      bRow[j] = Whh + (size_t)(part * H_ + n0 + j * 16 + lm) * H_;

    for (int kt = 0; kt < H_; kt += 32) {
      const bf16x16 a0 = load_frag(aRow0, kt, half);
      const bf16x16 a1 = load_frag(aRow1, kt, half);
#pragma unroll
      for (int j = 0; j < 4; ++j) {
        const bf16x16 b = load_frag(bRow[j], kt, half);
        acc[0][j] = __builtin_amdgcn_wmma_f32_16x16x32_bf16(
            false, a0, false, b, (short)0, acc[0][j], false, false);
        acc[1][j] = __builtin_amdgcn_wmma_f32_16x16x32_bf16(
            false, a1, false, b, (short)0, acc[1][j], false, false);
      }
    }
  }

  // ---- input contribution (parts r, z, inn) : K = 96 = 3 chunks ----
  if (part != 2) {
    const int gate = (part == 3) ? 2 : part;
    const __bf16* aRow0 = Enc + (size_t)(m0 + lm) * (T_ * I_) + t * I_;
    const __bf16* aRow1 = Enc + (size_t)(m0 + 16 + lm) * (T_ * I_) + t * I_;
    const __bf16* bRow[4];
#pragma unroll
    for (int j = 0; j < 4; ++j)
      bRow[j] = Wih + (size_t)(gate * H_ + n0 + j * 16 + lm) * I_;

#pragma unroll
    for (int kt = 0; kt < I_; kt += 32) {
      const bf16x16 a0 = load_frag(aRow0, kt, half);
      const bf16x16 a1 = load_frag(aRow1, kt, half);
#pragma unroll
      for (int j = 0; j < 4; ++j) {
        const bf16x16 b = load_frag(bRow[j], kt, half);
        acc[0][j] = __builtin_amdgcn_wmma_f32_16x16x32_bf16(
            false, a0, false, b, (short)0, acc[0][j], false, false);
        acc[1][j] = __builtin_amdgcn_wmma_f32_16x16x32_bf16(
            false, a1, false, b, (short)0, acc[1][j], false, false);
      }
    }
  }

  // ---- epilogue: bias (+ sigmoid for r,z), C-layout row = r + 8*half, col = lane%16
#pragma unroll
  for (int i = 0; i < 2; ++i) {
#pragma unroll
    for (int j = 0; j < 4; ++j) {
      const int col = n0 + j * 16 + lm;
      float bias;
      float* dst;
      if (part == 0)      { bias = b_ih[col] + b_hh[col];                   dst = Rg;  }
      else if (part == 1) { bias = b_ih[H_ + col] + b_hh[H_ + col];         dst = Zg;  }
      else if (part == 2) { bias = b_hh[2 * H_ + col];                      dst = HN;  }
      else                { bias = b_ih[2 * H_ + col];                      dst = INN; }
#pragma unroll
      for (int r = 0; r < 8; ++r) {
        const int row = m0 + i * 16 + r + 8 * half;
        float v = acc[i][j][r] + bias;
        if (part < 2) v = 1.0f / (1.0f + __expf(-v));
        dst[(size_t)row * H_ + col] = v;
      }
    }
  }
}

// Elementwise GRU update: h = (1-z)*tanh(inn + r*hn) + z*h (in place, + bf16 mirror)
__global__ __launch_bounds__(256)
void gru_update(const float* __restrict__ Rg, const float* __restrict__ Zg,
                const float* __restrict__ HN, const float* __restrict__ INN,
                float* __restrict__ Hf, __bf16* __restrict__ Hb,
                float* __restrict__ out) {
  const int idx = blockIdx.x * blockDim.x + threadIdx.x;
  if (idx >= B_ * H_) return;
  const float r = Rg[idx];
  const float z = Zg[idx];
  const float n = tanhf(INN[idx] + r * HN[idx]);
  const float h = Hf[idx];
  const float hnew = (1.0f - z) * n + z * h;
  Hf[idx] = hnew;
  Hb[idx] = (__bf16)hnew;
  if (out) out[idx] = hnew;
}

// One-shot prep: fp32 -> bf16 weight/input conversion, zero h.
__global__ __launch_bounds__(256)
void gru_prep(const float* __restrict__ enc, const float* __restrict__ wih,
              const float* __restrict__ whh, __bf16* __restrict__ encb,
              __bf16* __restrict__ wihb, __bf16* __restrict__ whhb,
              float* __restrict__ hf, __bf16* __restrict__ hb) {
  const int stride = gridDim.x * blockDim.x;
  const int i0 = blockIdx.x * blockDim.x + threadIdx.x;
  for (int i = i0; i < G3_ * H_; i += stride) whhb[i] = (__bf16)whh[i];
  for (int i = i0; i < G3_ * I_; i += stride) wihb[i] = (__bf16)wih[i];
  for (int i = i0; i < B_ * T_ * I_; i += stride) encb[i] = (__bf16)enc[i];
  for (int i = i0; i < B_ * H_; i += stride) { hf[i] = 0.0f; hb[i] = (__bf16)0.0f; }
}

extern "C" void kernel_launch(void* const* d_in, const int* in_sizes, int n_in,
                              void* d_out, int out_size, void* d_ws, size_t ws_size,
                              hipStream_t stream) {
  const float* enc = (const float*)d_in[0];  // [512,49,96]
  const float* wih = (const float*)d_in[1];  // [3072,96]
  const float* whh = (const float*)d_in[2];  // [3072,1024]
  const float* bih = (const float*)d_in[3];  // [3072]
  const float* bhh = (const float*)d_in[4];  // [3072]
  (void)in_sizes; (void)n_in; (void)out_size; (void)ws_size;

  char* ws = (char*)d_ws;
  size_t off = 0;
  __bf16* whhb = (__bf16*)(ws + off); off += (size_t)G3_ * H_ * 2;        // 6.29 MB
  __bf16* wihb = (__bf16*)(ws + off); off += (size_t)G3_ * I_ * 2;        // 0.59 MB
  __bf16* encb = (__bf16*)(ws + off); off += (size_t)B_ * T_ * I_ * 2;    // 4.82 MB
  float*  hf   = (float*)(ws + off);  off += (size_t)B_ * H_ * 4;         // 2 MB
  __bf16* hb   = (__bf16*)(ws + off); off += (size_t)B_ * H_ * 2;         // 1 MB
  float*  Rg   = (float*)(ws + off);  off += (size_t)B_ * H_ * 4;
  float*  Zg   = (float*)(ws + off);  off += (size_t)B_ * H_ * 4;
  float*  HN   = (float*)(ws + off);  off += (size_t)B_ * H_ * 4;
  float*  INN  = (float*)(ws + off);  off += (size_t)B_ * H_ * 4;         // total ~22 MB

  gru_prep<<<1024, 256, 0, stream>>>(enc, wih, whh, encb, wihb, whhb, hf, hb);

  for (int t = 0; t < T_; ++t) {
    gru_gate_gemm<<<dim3(8, 16), 256, 0, stream>>>(whhb, wihb, encb, hb,
                                                   bih, bhh, Rg, Zg, HN, INN, t);
    gru_update<<<(B_ * H_ + 255) / 256, 256, 0, stream>>>(
        Rg, Zg, HN, INN, hf, hb, (t == T_ - 1) ? (float*)d_out : nullptr);
  }
}
